// RelativePositionAttention_49787260895466
// MI455X (gfx1250) — matrix operand order repaired and checked by reading, not compiled, not run on hardware
//
#include <hip/hip_runtime.h>

#define BATCH  2
#define SEQ    2048
#define DMODEL 1024
#define NH     16
#define DKH    64
#define MRPC   128
#define NREL   257   // 2*MRPC + 1
#define RELPAD 288   // padded rel rows (9 chunks of K=32)
#define SROWP  (SEQ + 4)   // LDS score row pitch (floats), bank-conflict pad

typedef __attribute__((ext_vector_type(16))) __bf16 v16bf;
typedef __attribute__((ext_vector_type(8)))  __bf16 v8bf;
typedef __attribute__((ext_vector_type(4)))  __bf16 v4bf;
typedef __attribute__((ext_vector_type(8)))  float  v8f;
typedef __attribute__((ext_vector_type(4)))  float  v4f;

// ---------- helpers ----------
__device__ __forceinline__ v8f wmma_bf16(v16bf a, v16bf b, v8f c) {
  // D = A(16x32 bf16) * B(32x16 bf16) + C(16x16 f32)
  return __builtin_amdgcn_wmma_f32_16x16x32_bf16(false, a, false, b, (short)0, c,
                                                 false, false);
}

__device__ __forceinline__ v16bf packA(v4f x0, v4f x1, v4f x2, v4f x3) {
  v8bf lo = __builtin_shufflevector(__builtin_convertvector(x0, v4bf),
                                    __builtin_convertvector(x1, v4bf),
                                    0, 1, 2, 3, 4, 5, 6, 7);
  v8bf hi = __builtin_shufflevector(__builtin_convertvector(x2, v4bf),
                                    __builtin_convertvector(x3, v4bf),
                                    0, 1, 2, 3, 4, 5, 6, 7);
  return __builtin_shufflevector(lo, hi, 0, 1, 2, 3, 4, 5, 6, 7,
                                 8, 9, 10, 11, 12, 13, 14, 15);
}
// A-fragment (per-lane pattern: k+0..7 and k+16..23) from fp32 source
__device__ __forceinline__ v16bf loadA(const float* __restrict__ p) {
  return packA(*(const v4f*)(p), *(const v4f*)(p + 4),
               *(const v4f*)(p + 16), *(const v4f*)(p + 20));
}
// A-fragment from bf16 source
__device__ __forceinline__ v16bf loadA(const __bf16* __restrict__ p) {
  v8bf lo = *(const v8bf*)(p);
  v8bf hi = *(const v8bf*)(p + 16);
  return __builtin_shufflevector(lo, hi, 0, 1, 2, 3, 4, 5, 6, 7,
                                 8, 9, 10, 11, 12, 13, 14, 15);
}
// B-fragment (per-lane pattern: k+0..7 and k+16..23 of one column)
__device__ __forceinline__ v16bf loadB(const __bf16* __restrict__ p) {
  v8bf lo = *(const v8bf*)(p);
  v8bf hi = *(const v8bf*)(p + 16);
  return __builtin_shufflevector(lo, hi, 0, 1, 2, 3, 4, 5, 6, 7,
                                 8, 9, 10, 11, 12, 13, 14, 15);
}

// ---------- prep kernels ----------
__global__ __launch_bounds__(256)
void cvt_f32_bf16(const float* __restrict__ src, __bf16* __restrict__ dst, int n) {
  for (int i = blockIdx.x * 256 + threadIdx.x; i < n; i += gridDim.x * 256)
    dst[i] = (__bf16)src[i];
}
__global__ __launch_bounds__(256)
void pad_rel(const float* __restrict__ src, __bf16* __restrict__ dst) {
  int i = blockIdx.x * 256 + threadIdx.x;   // [RELPAD][DKH] row-major
  if (i < RELPAD * DKH) {
    int r = i >> 6;
    dst[i] = (__bf16)(r < NREL ? src[i] : 0.0f);
  }
}
__global__ __launch_bounds__(256)
void pad_rel_t(const float* __restrict__ src, __bf16* __restrict__ dst) {
  int i = blockIdx.x * 256 + threadIdx.x;   // [DKH][RELPAD] transposed
  if (i < DKH * RELPAD) {
    int dv = i / RELPAD, r = i % RELPAD;
    dst[i] = (__bf16)(r < NREL ? src[r * DKH + dv] : 0.0f);
  }
}

// =====================================================================
// GEMM: out[m][n] = sum_k A[m][k] * W[n][k] + bias[n]   (W already bf16)
// Block = 4 waves; each wave owns a 16(M) x 64(N) strip: A-frag reused 4x.
// OUTMODE 0: fp32 row-major (out-proj)
// OUTMODE 1: bf16 head-major  ((b*NH+h)*SEQ+s)*DKH+dk   (Q, K)
// OUTMODE 2: bf16 head-major transposed ((b*NH+h)*DKH+dk)*SEQ+s  (V)
// =====================================================================
template <typename TIN, int OUTMODE>
__global__ __launch_bounds__(128)
void wmma_gemm_nt(const TIN* __restrict__ A, const __bf16* __restrict__ W,
                  const float* __restrict__ bias, void* __restrict__ outv,
                  int M, int N, int K) {
  const int lane = threadIdx.x & 31;
  const int wave = threadIdx.x >> 5;
  const int m0 = blockIdx.y * 64 + wave * 16;
  const int n0 = blockIdx.x * 64;
  const int kbA = (lane >> 4) << 3;   // 0 or 8
  const int kbB = (lane >> 4) << 3;   // 0 or 8
  const int mlo = lane & 15;

  const TIN*    aRow = A + (size_t)(m0 + mlo) * K;
  const __bf16* wRow = W + (size_t)(n0 + mlo) * K;

  v8f acc[4] = {v8f{}, v8f{}, v8f{}, v8f{}};
  for (int k0 = 0; k0 < K; k0 += 32) {
    v16bf af = loadA(aRow + k0 + kbA);
#pragma unroll
    for (int nt = 0; nt < 4; ++nt) {
      v16bf bf = loadB(wRow + (size_t)nt * 16 * K + k0 + kbB);
      acc[nt] = wmma_bf16(af, bf, acc[nt]);
    }
  }

  const int mhi = (lane >> 4) << 3;
#pragma unroll
  for (int nt = 0; nt < 4; ++nt) {
    const int n = n0 + nt * 16 + mlo;
    const float bn = bias[n];
#pragma unroll
    for (int r = 0; r < 8; ++r) {
      int m = m0 + mhi + r;
      float v = acc[nt][r] + bn;
      if (OUTMODE == 0) {
        ((float*)outv)[(size_t)m * N + n] = v;
      } else {
        int b = m / SEQ, s = m % SEQ;
        int h = n / DKH, dk = n % DKH;
        size_t idx = (OUTMODE == 1)
            ? (((size_t)b * NH + h) * SEQ + s) * DKH + dk
            : (((size_t)b * NH + h) * DKH + dk) * SEQ + s;
        ((__bf16*)outv)[idx] = (__bf16)v;
      }
    }
  }
}

// =====================================================================
// Fused attention: scores (+rel-key) -> softmax -> attn write-once ->
// Arel built in closed form from LDS -> context (attn@V + Arel@rel_v).
// Block = 4 waves, one (b, h, 16-query tile). Dynamic LDS ~151 KB.
// =====================================================================
__global__ __launch_bounds__(128)
void attn_fused_kernel(const __bf16* __restrict__ Qh, const __bf16* __restrict__ Kh,
                       const __bf16* __restrict__ Vt, const __bf16* __restrict__ relKp,
                       const __bf16* __restrict__ relVt,
                       float* __restrict__ attn, __bf16* __restrict__ ctx) {
  extern __shared__ float smem[];
  float* Srow   = smem;                         // [16][SROWP]
  float* Qrel   = Srow + 16 * SROWP;            // [16][RELPAD] (reused as Arel)
  float* ctxAcc = Qrel + 16 * RELPAD;           // [16][DKH]
  float* red    = ctxAcc + 16 * DKH;            // [16][8] x2 halves

  const int q0 = blockIdx.x * 16;
  const int h  = blockIdx.y;
  const int b  = blockIdx.z;
  const int wave = threadIdx.x >> 5;
  const int lane = threadIdx.x & 31;
  const int kbA = (lane >> 4) << 3;
  const int kbB = (lane >> 4) << 3;
  const int mlo = lane & 15;
  const int mhi = (lane >> 4) << 3;

  for (int i = threadIdx.x; i < 16 * RELPAD; i += 128) Qrel[i] = 0.0f;
  for (int i = threadIdx.x; i < 16 * DKH; i += 128) ctxAcc[i] = 0.0f;
  __syncthreads();

  const size_t bh = (size_t)b * NH + h;
  const __bf16* Qp = Qh + bh * SEQ * DKH;
  const __bf16* Kp = Kh + bh * SEQ * DKH;

  // Q A-fragments (DK=64 -> two K=32 chunks)
  v16bf qa0, qa1;
  {
    const __bf16* qr = Qp + (size_t)(q0 + mlo) * DKH;
    qa0 = loadA(qr + kbA);
    qa1 = loadA(qr + 32 + kbA);
  }

  // Qrel = Q_tile @ relK^T : 17 column tiles of 16 cover the 257 buckets
  for (int t = wave; t < 17; t += 4) {
    const int col = t * 16 + mlo;
    const __bf16* rr = relKp + (size_t)col * DKH;
    v8f acc = {};
    acc = wmma_bf16(qa0, loadB(rr + kbB), acc);
    acc = wmma_bf16(qa1, loadB(rr + 32 + kbB), acc);
#pragma unroll
    for (int r = 0; r < 8; ++r) Qrel[(mhi + r) * RELPAD + col] = acc[r];
  }
  __syncthreads();

  // ---- scores into LDS ----
  for (int kt = wave; kt < SEQ / 16; kt += 4) {
    const int k0 = kt * 16;
    const __bf16* kr = Kp + (size_t)(k0 + mlo) * DKH;
    v8f acc = {};
    acc = wmma_bf16(qa0, loadB(kr + kbB), acc);
    acc = wmma_bf16(qa1, loadB(kr + 32 + kbB), acc);
#pragma unroll
    for (int r = 0; r < 8; ++r) {
      int m = mhi + r;
      int d = (k0 + mlo) - (q0 + m);
      d = d < -MRPC ? -MRPC : (d > MRPC ? MRPC : d);
      Srow[m * SROWP + k0 + mlo] =
          (acc[r] + Qrel[m * RELPAD + d + MRPC]) * 0.125f;
    }
  }
  __syncthreads();

  // ---- softmax in LDS; write normalized attn to HBM once ----
  const int row = threadIdx.x >> 3;   // 16 rows
  const int tr  = threadIdx.x & 7;    // 8 threads/row, 256 elems each
  {
    float* sr = Srow + row * SROWP + tr * 256;
    float mx = -3.4e38f;
    for (int j = 0; j < 64; ++j) {
      v4f v = *(const v4f*)(sr + j * 4);
      mx = fmaxf(mx, fmaxf(fmaxf(v[0], v[1]), fmaxf(v[2], v[3])));
    }
    red[row * 8 + tr] = mx;
    __syncthreads();
    mx = red[row * 8];
#pragma unroll
    for (int i = 1; i < 8; ++i) mx = fmaxf(mx, red[row * 8 + i]);
    __syncthreads();  // before reusing red for sums

    float sum = 0.0f;
    for (int j = 0; j < 64; ++j) {
      v4f v = *(const v4f*)(sr + j * 4);
      v[0] = __expf(v[0] - mx); v[1] = __expf(v[1] - mx);
      v[2] = __expf(v[2] - mx); v[3] = __expf(v[3] - mx);
      *(v4f*)(sr + j * 4) = v;
      sum += (v[0] + v[1]) + (v[2] + v[3]);
    }
    red[row * 8 + tr] = sum;
    __syncthreads();
    sum = 0.0f;
#pragma unroll
    for (int i = 0; i < 8; ++i) sum += red[row * 8 + i];
    const float inv = 1.0f / sum;

    float* gout = attn + (bh * SEQ + q0 + row) * SEQ + tr * 256;
    for (int j = 0; j < 64; ++j) {
      v4f v = *(const v4f*)(sr + j * 4);
      v *= inv;
      *(v4f*)(sr + j * 4) = v;      // keep normalized copy for context
      *(v4f*)(gout + j * 4) = v;    // the attn output (write-once)
    }
  }
  __syncthreads();

  // ---- Arel in closed form (reuse Qrel buffer) ----
  // bucket 0 = sum_{k<=q-128}; bucket 256 = sum_{k>=q+128};
  // bucket r in [1,255] = attn[q][q+r-128] (single element or 0).
  float* Arel = Qrel;
  {
    const int q = q0 + row;
    const float* srow = Srow + row * SROWP;
    float sumLow = 0.0f, sumHigh = 0.0f;
    const int endLow = q - MRPC + 1;            // k < endLow  (k <= q-128)
    for (int k = tr; k < endLow; k += 8) sumLow += srow[k];
    for (int k = q + MRPC + tr; k < SEQ; k += 8) sumHigh += srow[k];
    red[row * 8 + tr] = sumLow;
    red[128 + row * 8 + tr] = sumHigh;
    __syncthreads();
    float totLow = 0.0f, totHigh = 0.0f;
#pragma unroll
    for (int i = 0; i < 8; ++i) {
      totLow  += red[row * 8 + i];
      totHigh += red[128 + row * 8 + i];
    }
    for (int r = tr; r < RELPAD; r += 8) {
      float v;
      if (r >= NREL) v = 0.0f;
      else if (r == 0) v = totLow;
      else if (r == NREL - 1) v = totHigh;
      else {
        int k = q + r - MRPC;
        v = ((unsigned)k < (unsigned)SEQ) ? srow[k] : 0.0f;
      }
      Arel[row * RELPAD + r] = v;
    }
  }
  __syncthreads();

  // ---- context: attn@V (attn from LDS, V transposed) ----
  const __bf16* Vb = Vt + bh * DKH * SEQ;
  const float* srm = Srow + mlo * SROWP;
  v8f acc4[4] = {v8f{}, v8f{}, v8f{}, v8f{}};
  for (int c = wave; c < SEQ / 32; c += 4) {
    const int k0 = c * 32;
    v16bf af = packA(*(const v4f*)(srm + k0 + kbA),
                     *(const v4f*)(srm + k0 + kbA + 4),
                     *(const v4f*)(srm + k0 + kbA + 16),
                     *(const v4f*)(srm + k0 + kbA + 20));
#pragma unroll
    for (int nt = 0; nt < 4; ++nt) {
      const __bf16* vp = Vb + (size_t)(nt * 16 + mlo) * SEQ + k0 + kbB;
      acc4[nt] = wmma_bf16(af, loadB(vp), acc4[nt]);
    }
  }
#pragma unroll
  for (int nt = 0; nt < 4; ++nt) {
    const int n = nt * 16 + mlo;
#pragma unroll
    for (int r = 0; r < 8; ++r)
      atomicAdd(&ctxAcc[(mhi + r) * DKH + n], acc4[nt][r]);
  }
  __syncthreads();

  // ---- Arel @ rel_v (transposed rel_v: contiguous B loads) ----
  v8f racc[4] = {v8f{}, v8f{}, v8f{}, v8f{}};
  for (int c = wave; c < RELPAD / 32; c += 4) {   // 9 chunks, every wave >=1
    const int k0 = c * 32;
    const float* arl = Arel + mlo * RELPAD + k0 + kbA;
    v16bf af = packA(*(const v4f*)(arl), *(const v4f*)(arl + 4),
                     *(const v4f*)(arl + 16), *(const v4f*)(arl + 20));
#pragma unroll
    for (int nt = 0; nt < 4; ++nt) {
      const __bf16* rp = relVt + (size_t)(nt * 16 + mlo) * RELPAD + k0 + kbB;
      racc[nt] = wmma_bf16(af, loadB(rp), racc[nt]);
    }
  }
#pragma unroll
  for (int nt = 0; nt < 4; ++nt) {
    const int n = nt * 16 + mlo;
#pragma unroll
    for (int r = 0; r < 8; ++r)
      atomicAdd(&ctxAcc[(mhi + r) * DKH + n], racc[nt][r]);
  }
  __syncthreads();

  // ---- write ctx bf16, (b, s, h, dk) layout for the final GEMM ----
  for (int i = threadIdx.x; i < 16 * DKH; i += 128) {
    int m = i >> 6, dv = i & 63;
    ctx[(((size_t)b * SEQ + q0 + m) * NH + h) * DKH + dv] = (__bf16)ctxAcc[i];
  }
}

// =====================================================================
extern "C" void kernel_launch(void* const* d_in, const int* in_sizes, int n_in,
                              void* d_out, int out_size, void* d_ws, size_t ws_size,
                              hipStream_t stream) {
  (void)in_sizes; (void)n_in; (void)out_size; (void)ws_size;
  const float* query = (const float*)d_in[0];
  const float* key   = (const float*)d_in[1];
  const float* value = (const float*)d_in[2];
  const float* Wq = (const float*)d_in[3];  const float* bq = (const float*)d_in[4];
  const float* Wk = (const float*)d_in[5];  const float* bk = (const float*)d_in[6];
  const float* Wv = (const float*)d_in[7];  const float* bv = (const float*)d_in[8];
  const float* Wo = (const float*)d_in[9];  const float* bo = (const float*)d_in[10];
  const float* rel_k = (const float*)d_in[11];
  const float* rel_v = (const float*)d_in[12];

  const size_t elems = (size_t)BATCH * SEQ * DMODEL;   // 4,194,304
  const size_t welems = (size_t)DMODEL * DMODEL;       // 1,048,576
  __bf16* Qh    = (__bf16*)d_ws;
  __bf16* Kh    = Qh + elems;
  __bf16* Vt    = Kh + elems;           // V transposed head-major (b,h,dk,s)
  __bf16* ctx   = Vt + elems;
  __bf16* Wqb   = ctx + elems;
  __bf16* Wkb   = Wqb + welems;
  __bf16* Wvb   = Wkb + welems;
  __bf16* Wob   = Wvb + welems;
  __bf16* relKp = Wob + welems;                         // [RELPAD][DKH]
  __bf16* relVt = relKp + (size_t)RELPAD * DKH;         // [DKH][RELPAD]

  float* out  = (float*)d_out;
  float* attn = out + elems;   // (B,H,S,S) region of d_out

  // 0) prep: weights -> bf16, rel tables -> padded bf16 (rel_v transposed)
  cvt_f32_bf16<<<2048, 256, 0, stream>>>(Wq, Wqb, (int)welems);
  cvt_f32_bf16<<<2048, 256, 0, stream>>>(Wk, Wkb, (int)welems);
  cvt_f32_bf16<<<2048, 256, 0, stream>>>(Wv, Wvb, (int)welems);
  cvt_f32_bf16<<<2048, 256, 0, stream>>>(Wo, Wob, (int)welems);
  pad_rel  <<<(RELPAD * DKH + 255) / 256, 256, 0, stream>>>(rel_k, relKp);
  pad_rel_t<<<(DKH * RELPAD + 255) / 256, 256, 0, stream>>>(rel_v, relVt);

  const int M = BATCH * SEQ;               // 4096
  dim3 gGemm(DMODEL / 64, M / 64);         // (16, 64), 128 thr = 4 waves

  // 1) QKV projections (V written transposed for the context phase)
  wmma_gemm_nt<float, 1><<<gGemm, 128, 0, stream>>>(query, Wqb, bq, (void*)Qh, M, DMODEL, DMODEL);
  wmma_gemm_nt<float, 1><<<gGemm, 128, 0, stream>>>(key,   Wkb, bk, (void*)Kh, M, DMODEL, DMODEL);
  wmma_gemm_nt<float, 2><<<gGemm, 128, 0, stream>>>(value, Wvb, bv, (void*)Vt, M, DMODEL, DMODEL);

  // 2) fused scores + softmax + Arel + context; attn written once
  dim3 gAtt(SEQ / 16, NH, BATCH);          // (128, 16, 2)
  const size_t ldsBytes = (size_t)(16 * SROWP + 16 * RELPAD + 16 * DKH + 256)
                          * sizeof(float); // ~151 KB
  attn_fused_kernel<<<gAtt, 128, ldsBytes, stream>>>(Qh, Kh, Vt, relKp, relVt,
                                                     attn, ctx);

  // 3) output projection -> fp32 out
  wmma_gemm_nt<__bf16, 0><<<gGemm, 128, 0, stream>>>(ctx, Wob, bo, (void*)out, M, DMODEL, DMODEL);
}
